// AdaptiveRNNCell_35459249996054
// MI455X (gfx1250) — matrix operand order repaired and checked by reading, compiled, not run on hardware
//
#include <hip/hip_runtime.h>
#include <hip/hip_bf16.h>
#include <math.h>

typedef __bf16 bf16_t;
typedef __bf16 bf16x8  __attribute__((ext_vector_type(8)));
typedef __bf16 bf16x16 __attribute__((ext_vector_type(16)));
typedef float  f32x8   __attribute__((ext_vector_type(8)));

namespace act {
constexpr int B     = 8192;
constexpr int IN    = 1024;
constexpr int H     = 2048;
constexpr int H3    = 3 * H;
constexpr int TLIM  = 20;
constexpr int ROWS  = 64;            // batch rows per workgroup
constexpr int NWG   = B / ROWS;      // 128
constexpr int WAVES = 16;
constexpr int NTHR  = WAVES * 32;    // 512
constexpr int KT_H  = H / 32;        // 64 k-tiles (hidden GEMM)
constexpr int KT_IN = IN / 32;       // 32 k-tiles (input GEMM)
constexpr int CT_H  = H / 16;        // 128 hidden col-tiles
constexpr int CT_ALL= H3 / 16;       // 384 col-tiles over all gates
constexpr int LSTRIDE = H * 2 + 16;  // padded LDS row stride (bytes), bank-spread
constexpr int ISTRIDE = IN * 2 + 16;

// workspace byte offsets
constexpr size_t OFF_PACKHH = 0;
constexpr size_t SZ_PACKHH  = (size_t)CT_ALL * KT_H  * 32 * 16 * 2;  // 25,165,824
constexpr size_t OFF_PACKIH = OFF_PACKHH + SZ_PACKHH;
constexpr size_t SZ_PACKIH  = (size_t)CT_ALL * KT_IN * 32 * 16 * 2;  // 12,582,912
constexpr size_t OFF_XW     = OFF_PACKIH + SZ_PACKIH;
constexpr size_t SZ_XW      = (size_t)B * H3 * 4;                    // 201,326,592
constexpr size_t OFF_HSTAGE = OFF_XW + SZ_XW;
constexpr size_t SZ_HSTAGE  = (size_t)B * H * 2;                     // 33,554,432
constexpr size_t OFF_ACC    = OFF_HSTAGE + SZ_HSTAGE;

// accumulator float slots (in ws)
constexpr int A_CURVE = 0;   // 20 slots
constexpr int A_POND  = 20;
constexpr int A_RSUM  = 21;
constexpr int A_RSQ   = 22;
constexpr int A_NAT   = 23;
constexpr int A_FORC  = 24;
constexpr int A_HIST  = 32;  // 21 uint bins

// d_out element offsets (tuple flattened in return order)
constexpr size_t O_ACC   = 0;
constexpr size_t O_POND  = (size_t)B * H;
constexpr size_t O_STEPS = O_POND + 1;
constexpr size_t O_RMEAN = O_STEPS + B;
constexpr size_t O_RSTD  = O_RMEAN + 1;
constexpr size_t O_NAT   = O_RSTD + 1;
constexpr size_t O_FORC  = O_NAT + 1;
constexpr size_t O_P50   = O_FORC + 1;
constexpr size_t O_P90   = O_P50 + 1;
constexpr size_t O_CURVE = O_P90 + 1;
}

__device__ __forceinline__ float sigm(float x) { return 1.f / (1.f + __expf(-x)); }
__device__ __forceinline__ float tanh_fast(float x) {
    float t = __expf(-2.f * fabsf(x));
    float r = (1.f - t) / (1.f + t);
    return (x >= 0.f) ? r : -r;
}
__device__ __forceinline__ void atomF(float* p, float v) {
    __hip_atomic_fetch_add(p, v, __ATOMIC_RELAXED, __HIP_MEMORY_SCOPE_AGENT);
}
// A-fragment (16x32 bf16): one contiguous 32B run of 16 K per lane (two adjacent 16B chunks)
__device__ __forceinline__ bf16x16 joinA(const unsigned char* p) {
    bf16x8 lo = *(const bf16x8*)(p);
    bf16x8 hi = *(const bf16x8*)(p + 16);
    return __builtin_shufflevector(lo, hi, 0,1,2,3,4,5,6,7,8,9,10,11,12,13,14,15);
}
__device__ __forceinline__ f32x8 wmma_bf(bf16x16 a, bf16x16 b, f32x8 c) {
    return __builtin_amdgcn_wmma_f32_16x16x32_bf16(false, a, false, b, (short)0, c, false, false);
}

// ---- pack weights into B-fragment-ready bf16 layout (lane-contiguous 32B) ----
__global__ void act_pack_hh(const float* __restrict__ Whh, bf16_t* __restrict__ dst) {
    using namespace act;
    int gid = blockIdx.x * 256 + threadIdx.x;
    if (gid >= CT_ALL * KT_H * 32) return;
    int lane = gid & 31;
    int kt   = (gid >> 5) % KT_H;
    int ct   = gid / (32 * KT_H);
    int nrow = ct * 16 + (lane & 15);
    int k0   = kt * 32 + (lane >> 4) * 16;
    const float* s = Whh + (size_t)nrow * H + k0;
    bf16_t* d = dst + (size_t)gid * 16;
#pragma unroll
    for (int j = 0; j < 16; ++j) d[j] = (bf16_t)s[j];
}
__global__ void act_pack_ih(const float* __restrict__ Wih, bf16_t* __restrict__ dst) {
    using namespace act;
    int gid = blockIdx.x * 256 + threadIdx.x;
    if (gid >= CT_ALL * KT_IN * 32) return;
    int lane = gid & 31;
    int kt   = (gid >> 5) % KT_IN;
    int ct   = gid / (32 * KT_IN);
    int nrow = ct * 16 + (lane & 15);
    int k0   = kt * 32 + (lane >> 4) * 16;
    const float* s = Wih + (size_t)nrow * (IN + 1) + k0;   // skip flag column
    bf16_t* d = dst + (size_t)gid * 16;
#pragma unroll
    for (int j = 0; j < 16; ++j) d[j] = (bf16_t)s[j];
}
__global__ void act_zero(float* accW) { if (threadIdx.x < 64) accW[threadIdx.x] = 0.f; }

// ---- main persistent ACT-GRU kernel: one WG owns 64 batch rows for all steps ----
__global__ __launch_bounds__(act::NTHR, 1)
void act_main(const float* __restrict__ inp, const float* __restrict__ Wih,
              const float* __restrict__ bih, const float* __restrict__ bhh,
              const float* __restrict__ hwv, const float* __restrict__ hbp,
              const bf16_t* __restrict__ packHH, const bf16_t* __restrict__ packIH,
              float* __restrict__ xW, bf16_t* __restrict__ hStage,
              float* __restrict__ accW, float* __restrict__ accState,
              float* __restrict__ outSteps) {
    using namespace act;
    extern __shared__ unsigned char smem[];
    unsigned char* hBuf = smem;                                  // 64*4112 B
    float* haltLDS = (float*)(smem + (size_t)ROWS * LSTRIDE);    // [WAVES][ROWS]
    float* sAccP   = haltLDS + WAVES * ROWS;
    float* sPonder = sAccP + ROWS;
    float* sSteps  = sPonder + ROWS;
    float* sRem    = sSteps + ROWS;
    float* sNat    = sRem + ROWS;
    float* sForc   = sNat + ROWS;
    float* sPadj   = sForc + ROWS;
    int*   sStill  = (int*)(sPadj + ROWS);
    int*   sHist   = sStill + ROWS;   // 24 slots
    int*   sCnt    = sHist + 24;
    float* sSum    = (float*)(sCnt + 1);

    const int tid  = threadIdx.x;
    const int wv   = tid >> 5;
    const int lane = tid & 31;
    const int lm   = lane & 15;
    const int lh   = lane >> 4;
    const int row0 = blockIdx.x * ROWS;
    const float halt_b = hbp[0];

    // stage input tile to LDS as bf16
    for (int idx = tid; idx < ROWS * IN; idx += NTHR) {
        int r = idx >> 10, c = idx & (IN - 1);
        *(bf16_t*)(hBuf + r * ISTRIDE + c * 2) = (bf16_t)inp[(size_t)(row0 + r) * IN + c];
    }
    if (tid < ROWS) {
        sAccP[tid] = 0.f; sPonder[tid] = 0.f; sSteps[tid] = 0.f; sRem[tid] = 0.f;
        sNat[tid] = 0.f; sForc[tid] = 0.f; sPadj[tid] = 0.f; sStill[tid] = 1;
    }
    __syncthreads();

    // pre-phase: xW = input @ Wih[:, :IN]^T + bias_ih  (each wave writes exactly
    // the column tiles it consumes later -> same-wave store->load ordering)
    for (int g = 0; g < 3; ++g) {
        for (int ci = 0; ci < 8; ++ci) {
            const int ctG = g * CT_H + wv * 8 + ci;
            f32x8 acc[4];
#pragma unroll
            for (int rt = 0; rt < 4; ++rt)
#pragma unroll
                for (int v = 0; v < 8; ++v) acc[rt][v] = 0.f;
            for (int k = 0; k < KT_IN; ++k) {
                bf16x16 bfrag = *(const bf16x16*)(packIH + (((size_t)ctG * KT_IN + k) * 32 + lane) * 16);
                const int ko = k * 64 + lh * 32;
#pragma unroll
                for (int rt = 0; rt < 4; ++rt) {
                    bf16x16 a = joinA(hBuf + (rt * 16 + lm) * ISTRIDE + ko);
                    acc[rt] = wmma_bf(a, bfrag, acc[rt]);
                }
            }
            const int col = ctG * 16 + lm;
            const float bi = bih[col];
#pragma unroll
            for (int rt = 0; rt < 4; ++rt)
#pragma unroll
                for (int v = 0; v < 8; ++v)
                    xW[(size_t)(row0 + rt * 16 + v + 8 * lh) * H3 + col] = acc[rt][v] + bi;
        }
    }
    __threadfence_block();
    __syncthreads();

    int n;
    for (n = 0; n < TLIM; ++n) {
        float haltAcc[4][8];
#pragma unroll
        for (int rt = 0; rt < 4; ++rt)
#pragma unroll
            for (int v = 0; v < 8; ++v) haltAcc[rt][v] = 0.f;

        for (int ci = 0; ci < 8; ++ci) {
            const int ct  = wv * 8 + ci;
            const int col = ct * 16 + lm;
            float wfR = 0.f, wfZ = 0.f, wfN = 0.f;
            if (n == 0) {                                  // flag * w_flag (last Wih column)
                wfR = Wih[(size_t)(col)         * (IN + 1) + IN];
                wfZ = Wih[(size_t)(col + H)     * (IN + 1) + IN];
                wfN = Wih[(size_t)(col + 2 * H) * (IN + 1) + IN];
            }
            f32x8 aR[4], aZ[4], aN[4];
#pragma unroll
            for (int rt = 0; rt < 4; ++rt)
#pragma unroll
                for (int v = 0; v < 8; ++v) {
                    const size_t rb = (size_t)(row0 + rt * 16 + v + 8 * lh) * H3;
                    aR[rt][v] = xW[rb + col] + wfR;        // gi_r as C init
                    aZ[rt][v] = xW[rb + H + col] + wfZ;    // gi_z as C init
                    aN[rt][v] = 0.f;                       // gh_n kept separate
                }
            if (n != 0) {                                  // hidden==0 at step 0: skip GEMM
                for (int k = 0; k < KT_H; ++k) {
                    bf16x16 bR = *(const bf16x16*)(packHH + (((size_t)(ct           ) * KT_H + k) * 32 + lane) * 16);
                    bf16x16 bZ = *(const bf16x16*)(packHH + (((size_t)(ct + CT_H    ) * KT_H + k) * 32 + lane) * 16);
                    bf16x16 bN = *(const bf16x16*)(packHH + (((size_t)(ct + 2 * CT_H) * KT_H + k) * 32 + lane) * 16);
                    const int ko = k * 64 + lh * 32;
#pragma unroll
                    for (int rt = 0; rt < 4; ++rt) {
                        bf16x16 a = joinA(hBuf + (rt * 16 + lm) * LSTRIDE + ko);
                        aR[rt] = wmma_bf(a, bR, aR[rt]);
                        aZ[rt] = wmma_bf(a, bZ, aZ[rt]);
                        aN[rt] = wmma_bf(a, bN, aN[rt]);
                    }
                }
            }
            const float bhR = bhh[col], bhZ = bhh[H + col], bhN = bhh[2 * H + col];
            const float hwc = hwv[col];
#pragma unroll
            for (int rt = 0; rt < 4; ++rt)
#pragma unroll
                for (int v = 0; v < 8; ++v) {
                    const int lrow = rt * 16 + v + 8 * lh;
                    float r   = sigm(aR[rt][v] + bhR);
                    float z   = sigm(aZ[rt][v] + bhZ);
                    float ghn = aN[rt][v] + bhN;
                    float gin = xW[(size_t)(row0 + lrow) * H3 + 2 * H + col] + wfN;
                    float nc  = tanh_fast(gin + r * ghn);
                    float hOld = (n == 0) ? 0.f
                               : (float)(*(const bf16_t*)(hBuf + lrow * LSTRIDE + col * 2));
                    float hNew = (1.f - z) * nc + z * hOld;
                    hStage[(size_t)(row0 + lrow) * H + col] = (bf16_t)hNew;
                    haltAcc[rt][v] += hwc * hNew;          // partial h . halt_w
                }
        }
        // reduce halt partials over the 16 lanes of each half, one write per row
#pragma unroll
        for (int rt = 0; rt < 4; ++rt)
#pragma unroll
            for (int v = 0; v < 8; ++v) {
                float s = haltAcc[rt][v];
                s += __shfl_xor(s, 1, 32);
                s += __shfl_xor(s, 2, 32);
                s += __shfl_xor(s, 4, 32);
                s += __shfl_xor(s, 8, 32);
                haltAcc[rt][v] = s;
            }
        if (lm == 0)
#pragma unroll
            for (int rt = 0; rt < 4; ++rt)
#pragma unroll
                for (int v = 0; v < 8; ++v)
                    haltLDS[wv * ROWS + rt * 16 + v + 8 * lh] = haltAcc[rt][v];
        __threadfence_block();
        __syncthreads();

        // ACT halting state machine (one thread per row)
        if (tid < ROWS) {
            float s = 0.f;
#pragma unroll
            for (int w = 0; w < WAVES; ++w) s += haltLDS[w * ROWS + tid];
            const float pt   = sigm(s + halt_b);
            const float accp = sAccP[tid];
            const int still  = sStill[tid];
            const int natm   = (accp + pt >= 0.99f) && still;
            const int forcm  = (n == TLIM - 1) && still && !natm;
            const int haltf  = natm | forcm;
            float padj = haltf ? (1.f - accp) : pt;
            padj = still ? padj : 0.f;
            const float rem = haltf ? (1.f - accp) : 0.f;
            const float sf  = still ? 1.f : 0.f;
            sPadj[tid]    = padj;
            sAccP[tid]    = accp + padj;
            sPonder[tid] += sf + rem;
            sSteps[tid]  += sf;
            sRem[tid]    += rem;
            sNat[tid]    += (float)natm;
            sForc[tid]   += (float)forcm;
            sStill[tid]   = still && !haltf;
        }
        __syncthreads();
        if (tid == 0) {
            float s = 0.f; int c = 0;
            for (int r = 0; r < ROWS; ++r) { s += sAccP[r]; c += sStill[r]; }
            atomF(accW + A_CURVE + n, s);
            *sSum = s; *sCnt = c;
        }
        // copy new hidden into LDS + accumulate acc_state (row uniform per j)
        for (int j = 0; j < (ROWS * H) / NTHR; ++j) {
            const int idx = j * NTHR + tid;
            const int r = idx >> 11, c = idx & (H - 1);
            const bf16_t h = hStage[(size_t)(row0 + r) * H + c];
            *(bf16_t*)(hBuf + r * LSTRIDE + c * 2) = h;
            const float p = sPadj[r];
            if (p != 0.f || n == 0) {
                const size_t o = (size_t)(row0 + r) * H + c;
                const float prev = (n == 0) ? 0.f : accState[o];
                accState[o] = prev + p * (float)h;
            }
        }
        __syncthreads();
        if (*sCnt == 0) { ++n; break; }   // all rows halted: remaining steps are unobservable
    }

    // tail: frozen acc_p contributes to the remaining curve points
    if (tid == 0) {
        for (int m = n; m < TLIM; ++m) atomF(accW + A_CURVE + m, *sSum);
        float pond = 0.f, rs = 0.f, rq = 0.f, na = 0.f, fo = 0.f;
        for (int r = 0; r < ROWS; ++r) {
            pond += sPonder[r]; rs += sRem[r]; rq += sRem[r] * sRem[r];
            na += sNat[r]; fo += sForc[r];
        }
        atomF(accW + A_POND, pond); atomF(accW + A_RSUM, rs); atomF(accW + A_RSQ, rq);
        atomF(accW + A_NAT, na);    atomF(accW + A_FORC, fo);
    }
    if (tid < ROWS) outSteps[row0 + tid] = sSteps[tid];
    if (tid < 24) sHist[tid] = 0;
    __syncthreads();
    if (tid < ROWS) {
        int b = (int)(sSteps[tid] + 0.5f);
        b = b < 0 ? 0 : (b > 20 ? 20 : b);
        atomicAdd(&sHist[b], 1);
    }
    __syncthreads();
    if (tid < 21 && sHist[tid] > 0)
        __hip_atomic_fetch_add((unsigned int*)(accW + A_HIST) + tid,
                               (unsigned int)sHist[tid],
                               __ATOMIC_RELAXED, __HIP_MEMORY_SCOPE_AGENT);
}

__device__ float quant_from_hist(const unsigned int* h, float q) {
    float pos = q * (float)(act::B - 1);
    int lo = (int)floorf(pos);
    float fr = pos - (float)lo;
    int cum = 0; float vlo = 20.f; bool found = false;
    for (int s = 0; s <= 20; ++s) {
        cum += (int)h[s];
        if (!found && cum > lo) { vlo = (float)s; found = true; }
        if (cum > lo + 1) return vlo + fr * ((float)s - vlo);
    }
    return vlo;
}

__global__ void act_finalize(const float* __restrict__ accW, float* __restrict__ out) {
    using namespace act;
    int t = threadIdx.x;
    if (t < TLIM) out[O_CURVE + t] = accW[A_CURVE + t] / (float)B;
    if (t == 0) {
        out[O_POND] = accW[A_POND] * (0.001f / (float)B);
        float rm = accW[A_RSUM] / (float)B;
        out[O_RMEAN] = rm;
        float var = accW[A_RSQ] / (float)B - rm * rm;
        out[O_RSTD] = sqrtf(fmaxf(var, 0.f));
        out[O_NAT]  = accW[A_NAT]  / (float)B;
        out[O_FORC] = accW[A_FORC] / (float)B;
        const unsigned int* hist = (const unsigned int*)(accW + A_HIST);
        out[O_P50] = quant_from_hist(hist, 0.5f);
        out[O_P90] = quant_from_hist(hist, 0.9f);
    }
}

extern "C" void kernel_launch(void* const* d_in, const int* in_sizes, int n_in,
                              void* d_out, int out_size, void* d_ws, size_t ws_size,
                              hipStream_t stream) {
    using namespace act;
    (void)in_sizes; (void)n_in; (void)out_size; (void)ws_size;
    const float* inp = (const float*)d_in[0];
    const float* Wih = (const float*)d_in[1];
    const float* Whh = (const float*)d_in[2];
    const float* bih = (const float*)d_in[3];
    const float* bhh = (const float*)d_in[4];
    const float* hw  = (const float*)d_in[5];
    const float* hb  = (const float*)d_in[6];
    float* out = (float*)d_out;
    unsigned char* ws = (unsigned char*)d_ws;
    bf16_t* packHH = (bf16_t*)(ws + OFF_PACKHH);
    bf16_t* packIH = (bf16_t*)(ws + OFF_PACKIH);
    float*  xW     = (float*) (ws + OFF_XW);
    bf16_t* hStage = (bf16_t*)(ws + OFF_HSTAGE);
    float*  accW   = (float*) (ws + OFF_ACC);

    act_zero<<<1, 64, 0, stream>>>(accW);
    act_pack_hh<<<(CT_ALL * KT_H  * 32 + 255) / 256, 256, 0, stream>>>(Whh, packHH);
    act_pack_ih<<<(CT_ALL * KT_IN * 32 + 255) / 256, 256, 0, stream>>>(Wih, packIH);

    size_t smemBytes = (size_t)ROWS * LSTRIDE + (size_t)WAVES * ROWS * 4
                     + 8 * ROWS * 4 + 24 * 4 + 16;   // ~269.5 KB of the 320 KB WGP LDS
    act_main<<<NWG, NTHR, smemBytes, stream>>>(inp, Wih, bih, bhh, hw, hb,
        packHH, packIH, xW, hStage, accW, out + O_ACC, out + O_STEPS);
    act_finalize<<<1, 64, 0, stream>>>(accW, out);
}
